// ConditionalPointFFM_1236950581343
// MI455X (gfx1250) — compile-verified, matrix operand order checked by reading, not run on hardware
//
#include <hip/hip_runtime.h>
#include <hip/hip_bf16.h>
#include <math.h>

// ---------------- problem constants (match reference) ----------------
#define NBATCH   4
#define NPTS     32768
#define NOBS     512
#define NFIELDS  4
#define HID      256
#define COND     128
#define FED      32
#define TOTAL_ROWS (NBATCH * NPTS)   // 131072
#define TM  128
#define TK  32

typedef __attribute__((ext_vector_type(16))) _Float16 v16h;
typedef __attribute__((ext_vector_type(8)))  _Float16 v8h;
typedef __attribute__((ext_vector_type(8)))  float    v8f;

__device__ __forceinline__ float gelu_exact(float x) {
    return 0.5f * x * (1.0f + erff(x * 0.70710678118654752f));
}

// k offset inside a 32-wide K-tile for 16-bit A/B fragments (ISA 7.12.2):
// lanes 0-15 cover K {0..7, 16..23}; lanes 16-31 cover K {8..15, 24..31}
__device__ __forceinline__ int wmma_kofs(int e, int lane) {
    return ((e < 8) ? e : e + 8) + ((lane & 16) ? 8 : 0);
}

__device__ __forceinline__ v16h load_fragA(const _Float16* p) {
    v8h lo = *(const v8h*)p;
    v8h hi = *(const v8h*)(p + 16);
    v16h a;
#pragma unroll
    for (int e = 0; e < 8; ++e) { a[e] = lo[e]; a[e + 8] = hi[e]; }
    return a;
}

// ---------------- prep kernels ----------------
// Swizzle W[K][N] (f32, row-major) into fragment-major f16:
// dst[((kt*(N/16)+nt)*32 + lane)*16 + e], zero-padded to Kp (multiple of 32).
__global__ void k_swizzle_W(const float* __restrict__ W, _Float16* __restrict__ dst,
                            int K, int N, int Kp) {
    int idx = blockIdx.x * 256 + threadIdx.x;
    int total = (Kp >> 5) * (N >> 4) * 512;
    if (idx >= total) return;
    int e    = idx & 15;
    int lane = (idx >> 4) & 31;
    int tile = idx >> 9;
    int nt   = tile % (N >> 4);
    int kt   = tile / (N >> 4);
    int k = kt * 32 + wmma_kofs(e, lane);
    int n = nt * 16 + (lane & 15);
    dst[idx] = (k < K) ? (_Float16)W[(size_t)k * N + n] : (_Float16)0.0f;
}

__global__ void k_zero_f32(float* __restrict__ p, int n) {
    int i = blockIdx.x * 256 + threadIdx.x;
    if (i < n) p[i] = 0.0f;
}

// build padded point-encoder input rows: [coords(3)|x_t(4)|t(1)|0...], stride 32
__global__ void k_build_in0(const float* __restrict__ coords, const float* __restrict__ x_t,
                            const float* __restrict__ t, _Float16* __restrict__ in0) {
    int r = blockIdx.x * 256 + threadIdx.x;
    if (r >= TOTAL_ROWS) return;
    int b = r / NPTS;
    const float* c = coords + (size_t)r * 3;
    const float* x = x_t + (size_t)r * 4;
    v8h a, z;
    a[0] = (_Float16)c[0]; a[1] = (_Float16)c[1]; a[2] = (_Float16)c[2];
    a[3] = (_Float16)x[0]; a[4] = (_Float16)x[1]; a[5] = (_Float16)x[2]; a[6] = (_Float16)x[3];
    a[7] = (_Float16)t[b];
#pragma unroll
    for (int e = 0; e < 8; ++e) z[e] = (_Float16)0.0f;
    v8h* o = (v8h*)(in0 + (size_t)r * 32);
    o[0] = a; o[1] = z; o[2] = z; o[3] = z;
}

// Precompute per-obs softmax constants (shared by stats + lc kernels):
//   s4x/s4y/s4z = 400*obs_coord (folds -d2/(2*sigma^2) scaling into the dot)
//   sb          = -200*|o|^2, or -2e32 when masked (=> exp underflows to 0)
__global__ void k_obs_pre(const float* __restrict__ obs_coords,
                          const float* __restrict__ obs_mask,
                          float* __restrict__ s4x, float* __restrict__ s4y,
                          float* __restrict__ s4z, float* __restrict__ sb) {
    int i = blockIdx.x * 256 + threadIdx.x;
    if (i >= NBATCH * NOBS) return;
    float x = obs_coords[(size_t)i * 3 + 0];
    float y = obs_coords[(size_t)i * 3 + 1];
    float z = obs_coords[(size_t)i * 3 + 2];
    s4x[i] = 400.0f * x; s4y[i] = 400.0f * y; s4z[i] = 400.0f * z;
    sb[i] = (obs_mask[i] > 0.0f) ? (-200.0f * (x * x + y * y + z * z)) : -2.0e32f;
}

// ---------------- obs encoder (tiny: 2048 rows); writes swizzled obs_feat ----------------
__global__ void k_obs_encoder(const float* __restrict__ obs_coords,
                              const float* __restrict__ obs_values,
                              const float* __restrict__ obs_mask,
                              const int*   __restrict__ obs_ids,
                              const float* __restrict__ fe,
                              const float* __restrict__ W0, const float* __restrict__ b0,
                              const float* __restrict__ W1, const float* __restrict__ b1,
                              const float* __restrict__ W2, const float* __restrict__ b2,
                              _Float16* __restrict__ ofswz) {
    int ob = blockIdx.x;          // 0 .. B*NOBS-1
    int j  = threadIdx.x;         // 0 .. 127  (N dim)
    __shared__ float sin[36];
    __shared__ float sh[COND];
    float m = obs_mask[ob];
    if (j < 3)  sin[j] = obs_coords[(size_t)ob * 3 + j];
    if (j == 3) sin[3] = obs_values[ob];
    if (j >= 4 && j < 36) {
        int id = obs_ids[ob]; if (id < 0) id = 0;
        sin[j] = fe[id * FED + (j - 4)] * m;
    }
    __syncthreads();
    float a = b0[j];
    for (int k = 0; k < 36; ++k) a = fmaf(sin[k], W0[k * COND + j], a);
    a = gelu_exact(a);
    sh[j] = a;
    __syncthreads();
    float a1 = b1[j];
    for (int k = 0; k < COND; ++k) a1 = fmaf(sh[k], W1[k * COND + j], a1);
    a1 = gelu_exact(a1);
    __syncthreads();
    sh[j] = a1;
    __syncthreads();
    float a2 = b2[j];
    for (int k = 0; k < COND; ++k) a2 = fmaf(sh[k], W2[k * COND + j], a2);
    // write directly in WMMA-fragment layout (obs index = K dim, j = N dim)
    int b  = ob / NOBS, ko = ob % NOBS;
    int kt = ko >> 5,  kr = ko & 31;
    int lane = (j & 15) + (((kr & 15) >= 8) ? 16 : 0);
    int e    = (kr < 16) ? (kr & 7) : ((kr & 7) + 8);
    size_t idx = (((size_t)b * (NOBS >> 5) + kt) * (COND >> 4) + (j >> 4)) * 512 + lane * 16 + e;
    ofswz[idx] = (_Float16)(a2 * m);
}

// ---------------- generic WMMA GEMM: C = act(A @ W + b) ----------------
// block: 256 threads (8 waves), tile 128x128, K step 32. Each wave owns a
// 32x64 sub-tile (2 M-frags x 4 N-frags) to minimize redundant fragment loads.
// ASRC==0: A row-major [M][Kp].  ASRC==1: concat(pf[256] | g-broadcast[256] | lc[128]).
template <int ASRC>
__device__ __forceinline__ const _Float16* a_ptr(const _Float16* A, const _Float16* Ag,
                                                 const _Float16* Alc, int arow,
                                                 int k0, int kh, int Kp) {
    if (ASRC == 0)         return A   + (size_t)arow * Kp + k0 + kh;
    if (k0 < HID)          return A   + (size_t)arow * HID + k0 + kh;
    if (k0 < 2 * HID)      return Ag  + (size_t)(arow / NPTS) * HID + (k0 - HID) + kh;
    return Alc + (size_t)arow * COND + (k0 - 2 * HID) + kh;
}

template <int ASRC, bool ACT, bool DOSUM>
__global__ void k_gemm(const _Float16* __restrict__ A,
                       const _Float16* __restrict__ Ag,
                       const _Float16* __restrict__ Alc,
                       const _Float16* __restrict__ Wswz,   // fragment-major
                       const float*    __restrict__ bias,
                       _Float16*       __restrict__ C,
                       float*          __restrict__ sumOut,
                       int M, int N, int Kp) {
    __shared__ float ssum[128];
    int row0 = blockIdx.x * TM;
    int col0 = blockIdx.y * 128;
    int lane = threadIdx.x & 31;
    int wv   = threadIdx.x >> 5;
    int mtg  = wv >> 1;            // 0..3 : which 32-row group
    int ntg  = wv & 1;             // 0..1 : which 64-col group
    int nt16 = N >> 4;

    if (DOSUM && threadIdx.x < 128) ssum[threadIdx.x] = 0.0f;

    int arow = row0 + mtg * 32 + (lane & 15);
    int kh   = (lane & 16) ? 8 : 0;

    v8f acc[2][4];
#pragma unroll
    for (int mi = 0; mi < 2; ++mi)
#pragma unroll
        for (int ni = 0; ni < 4; ++ni)
#pragma unroll
            for (int r = 0; r < 8; ++r) acc[mi][ni][r] = 0.0f;

    for (int kt = 0; kt < (Kp >> 5); ++kt) {
        int k0 = kt * 32;
        v16h a0 = load_fragA(a_ptr<ASRC>(A, Ag, Alc, arow,      k0, kh, Kp));
        v16h a1 = load_fragA(a_ptr<ASRC>(A, Ag, Alc, arow + 16, k0, kh, Kp));
        const _Float16* wp = Wswz +
            (((size_t)kt * nt16 + (col0 >> 4) + ntg * 4) * 32 + lane) * 16;
#pragma unroll
        for (int ni = 0; ni < 4; ++ni) {
            v16h bfr = *(const v16h*)(wp + (size_t)ni * 512);
            acc[0][ni] = __builtin_amdgcn_wmma_f32_16x16x32_f16(
                false, a0, false, bfr, (short)0, acc[0][ni], false, false);
            acc[1][ni] = __builtin_amdgcn_wmma_f32_16x16x32_f16(
                false, a1, false, bfr, (short)0, acc[1][ni], false, false);
        }
    }

#pragma unroll
    for (int mi = 0; mi < 2; ++mi)
#pragma unroll
        for (int ni = 0; ni < 4; ++ni) {
            int ncol = ntg * 64 + ni * 16 + (lane & 15);
            int col  = col0 + ncol;
            float bc = bias[col];
#pragma unroll
            for (int rI = 0; rI < 8; ++rI) {
                int row = row0 + mtg * 32 + mi * 16 + rI + ((lane & 16) ? 8 : 0);
                float v = acc[mi][ni][rI] + bc;
                if (ACT) v = gelu_exact(v);
                C[(size_t)row * N + col] = (_Float16)v;
                if (DOSUM) atomicAdd(&ssum[ncol], v);   // ds_add_f32
            }
        }
    if (DOSUM) {
        __syncthreads();
        if (threadIdx.x < 128) {
            int b = row0 / NPTS;
            atomicAdd(&sumOut[b * N + col0 + threadIdx.x], ssum[threadIdx.x]);
        }
    }
}

// ---------------- global encoder (tiny, B=4 rows) ----------------
__global__ void k_global_mlp(const float* __restrict__ sum_pf,
                             const float* __restrict__ W0, const float* __restrict__ b0,
                             const float* __restrict__ W1, const float* __restrict__ b1,
                             _Float16* __restrict__ g16) {
    int b = blockIdx.x, j = threadIdx.x;
    __shared__ float s0[HID], s1[HID];
    s0[j] = sum_pf[b * HID + j] * (1.0f / (float)NPTS);
    __syncthreads();
    float a = b0[j];
    for (int k = 0; k < HID; ++k) a = fmaf(s0[k], W0[k * HID + j], a);
    a = gelu_exact(a);
    s1[j] = a;
    __syncthreads();
    float a1 = b1[j];
    for (int k = 0; k < HID; ++k) a1 = fmaf(s1[k], W1[k * HID + j], a1);
    g16[b * HID + j] = (_Float16)a1;
}

// ---------------- softmax row stats (online logsumexp per query, branchless) ----------------
__global__ void k_softmax_stats(const float* __restrict__ coords,
                                const float* __restrict__ s4x, const float* __restrict__ s4y,
                                const float* __restrict__ s4z, const float* __restrict__ sb,
                                float* __restrict__ rmax, float* __restrict__ rinv) {
    __shared__ float sox[NOBS], soy[NOBS], soz[NOBS], sbb[NOBS];
    int r0 = blockIdx.x * 256;
    int b  = r0 / NPTS;
    for (int i = threadIdx.x; i < NOBS; i += 256) {
        sox[i] = s4x[b * NOBS + i];
        soy[i] = s4y[b * NOBS + i];
        soz[i] = s4z[b * NOBS + i];
        sbb[i] = sb[b * NOBS + i];
    }
    __syncthreads();
    int r = r0 + threadIdx.x;
    float qx = coords[(size_t)r * 3 + 0];
    float qy = coords[(size_t)r * 3 + 1];
    float qz = coords[(size_t)r * 3 + 2];
    float a0 = -200.0f * (qx * qx + qy * qy + qz * qz);
    float m = -3.4e38f, s = 0.0f;
    for (int i = 0; i < NOBS; ++i) {
        // logit = -max(d2,0)/(2 sigma^2) = min(a0 + sb + 400*q.o, 0)
        float l = a0 + sbb[i];
        l = fmaf(qx, sox[i], l);
        l = fmaf(qy, soy[i], l);
        l = fmaf(qz, soz[i], l);
        l = fminf(l, 0.0f);
        float mn = fmaxf(m, l);
        s = s * __expf(m - mn) + __expf(l - mn);
        m = mn;
    }
    rmax[r] = m;
    rinv[r] = 1.0f / s;
}

// ---------------- local_cond = softmax(w) @ obs_feat, w recomputed in A-frag ----------------
__global__ void k_lc_gemm(const float* __restrict__ coords,
                          const float* __restrict__ s4x, const float* __restrict__ s4y,
                          const float* __restrict__ s4z, const float* __restrict__ sb,
                          const float* __restrict__ rmax, const float* __restrict__ rinv,
                          const _Float16* __restrict__ ofswz,   // fragment-major per batch
                          _Float16* __restrict__ lc16) {
    __shared__ float sox[NOBS], soy[NOBS], soz[NOBS], sbb[NOBS];
    int row0 = blockIdx.x * 128;
    int b    = row0 / NPTS;
    for (int i = threadIdx.x; i < NOBS; i += 256) {
        sox[i] = s4x[b * NOBS + i];
        soy[i] = s4y[b * NOBS + i];
        soz[i] = s4z[b * NOBS + i];
        sbb[i] = sb[b * NOBS + i];
    }
    __syncthreads();

    int lane = threadIdx.x & 31;
    int wv   = threadIdx.x >> 5;
    int mb   = wv * 16;
    const _Float16* of = ofswz + (size_t)b * (NOBS * COND);

    v8f acc[8];
#pragma unroll
    for (int t = 0; t < 8; ++t)
#pragma unroll
        for (int r = 0; r < 8; ++r) acc[t][r] = 0.0f;

    int r = row0 + mb + (lane & 15);
    float qx = coords[(size_t)r * 3 + 0];
    float qy = coords[(size_t)r * 3 + 1];
    float qz = coords[(size_t)r * 3 + 2];
    float rm = rmax[r];
    float ri = rinv[r];
    float arm   = -200.0f * (qx * qx + qy * qy + qz * qz) - rm;   // a0 - rowmax
    float negrm = -rm;

    for (int k0 = 0; k0 < NOBS; k0 += TK) {
        v16h a;
#pragma unroll
        for (int e = 0; e < 16; ++e) {
            int k = k0 + wmma_kofs(e, lane);
            // min(logit,0) - rowmax  ==  min(arm + sb + 400*q.o, -rowmax)
            float l = arm + sbb[k];
            l = fmaf(qx, sox[k], l);
            l = fmaf(qy, soy[k], l);
            l = fmaf(qz, soz[k], l);
            l = fminf(l, negrm);
            a[e] = (_Float16)(__expf(l) * ri);
        }
        const _Float16* wp = of + (((size_t)(k0 >> 5) * (COND >> 4)) * 32 + lane) * 16;
#pragma unroll
        for (int nt = 0; nt < 8; ++nt) {
            v16h bfr = *(const v16h*)(wp + (size_t)nt * 512);
            acc[nt] = __builtin_amdgcn_wmma_f32_16x16x32_f16(
                false, a, false, bfr, (short)0, acc[nt], false, false);
        }
    }
#pragma unroll
    for (int nt = 0; nt < 8; ++nt) {
        int col = nt * 16 + (lane & 15);
#pragma unroll
        for (int rI = 0; rI < 8; ++rI) {
            int row = row0 + mb + rI + ((lane & 16) ? 8 : 0);
            lc16[(size_t)row * COND + col] = (_Float16)acc[nt][rI];
        }
    }
}

// ---------------- final head layer 256 -> 4 (tiny N, scalar; vectorized loads) ----------------
__global__ void k_head_out(const _Float16* __restrict__ h2,
                           const float* __restrict__ W2, const float* __restrict__ b2,
                           float* __restrict__ out) {
    int idx = blockIdx.x * 256 + threadIdx.x;   // row*4 + field
    if (idx >= TOTAL_ROWS * NFIELDS) return;
    int r = idx >> 2, f = idx & 3;
    const v8h* hv = (const v8h*)(h2 + (size_t)r * HID);
    float acc = b2[f];
    for (int c = 0; c < HID / 8; ++c) {
        v8h h8 = hv[c];
#pragma unroll
        for (int e = 0; e < 8; ++e) acc = fmaf((float)h8[e], W2[(c * 8 + e) * NFIELDS + f], acc);
    }
    out[idx] = acc;
}

// ---------------- host orchestration ----------------
extern "C" void kernel_launch(void* const* d_in, const int* in_sizes, int n_in,
                              void* d_out, int out_size, void* d_ws, size_t ws_size,
                              hipStream_t stream) {
    (void)in_sizes; (void)n_in; (void)out_size; (void)ws_size;
    const float* t_in       = (const float*)d_in[0];
    const float* x_t        = (const float*)d_in[1];
    const float* coords     = (const float*)d_in[2];
    const float* obs_coords = (const float*)d_in[3];
    const float* obs_values = (const float*)d_in[4];
    const float* obs_mask   = (const float*)d_in[5];
    const int*   obs_ids    = (const int*)d_in[6];
    const float* fe         = (const float*)d_in[7];
    const float* peW0 = (const float*)d_in[8];  const float* peB0 = (const float*)d_in[9];
    const float* peW1 = (const float*)d_in[10]; const float* peB1 = (const float*)d_in[11];
    const float* peW2 = (const float*)d_in[12]; const float* peB2 = (const float*)d_in[13];
    const float* oeW0 = (const float*)d_in[14]; const float* oeB0 = (const float*)d_in[15];
    const float* oeW1 = (const float*)d_in[16]; const float* oeB1 = (const float*)d_in[17];
    const float* oeW2 = (const float*)d_in[18]; const float* oeB2 = (const float*)d_in[19];
    const float* geW0 = (const float*)d_in[20]; const float* geB0 = (const float*)d_in[21];
    const float* geW1 = (const float*)d_in[22]; const float* geB1 = (const float*)d_in[23];
    const float* hdW0 = (const float*)d_in[24]; const float* hdB0 = (const float*)d_in[25];
    const float* hdW1 = (const float*)d_in[26]; const float* hdB1 = (const float*)d_in[27];
    const float* hdW2 = (const float*)d_in[28]; const float* hdB2 = (const float*)d_in[29];
    float* out = (float*)d_out;

    // ---- carve workspace ----
    char* w = (char*)d_ws;
    auto carve = [&](size_t bytes) { char* p = w; w += (bytes + 255) & ~(size_t)255; return p; };
    _Float16* peW0s = (_Float16*)carve((size_t)32 * HID * 2);            // K padded 8->32
    _Float16* peW1s = (_Float16*)carve((size_t)HID * HID * 2);
    _Float16* peW2s = (_Float16*)carve((size_t)HID * HID * 2);
    _Float16* hdW0s = (_Float16*)carve((size_t)(2 * HID + COND) * HID * 2);
    _Float16* hdW1s = (_Float16*)carve((size_t)HID * HID * 2);
    _Float16* in0   = (_Float16*)carve((size_t)TOTAL_ROWS * 32 * 2);     // padded stride 32
    _Float16* hA    = (_Float16*)carve((size_t)TOTAL_ROWS * HID * 2);    // ping
    _Float16* hB    = (_Float16*)carve((size_t)TOTAL_ROWS * HID * 2);    // pong
    _Float16* pf16  = (_Float16*)carve((size_t)TOTAL_ROWS * HID * 2);
    _Float16* lc16  = (_Float16*)carve((size_t)TOTAL_ROWS * COND * 2);
    _Float16* ofswz = (_Float16*)carve((size_t)NBATCH * NOBS * COND * 2);
    _Float16* g16   = (_Float16*)carve((size_t)NBATCH * HID * 2);
    float*    sumpf = (float*)carve((size_t)NBATCH * HID * 4);
    float*    rmax  = (float*)carve((size_t)TOTAL_ROWS * 4);
    float*    rinv  = (float*)carve((size_t)TOTAL_ROWS * 4);
    float*    s4x   = (float*)carve((size_t)NBATCH * NOBS * 4);
    float*    s4y   = (float*)carve((size_t)NBATCH * NOBS * 4);
    float*    s4z   = (float*)carve((size_t)NBATCH * NOBS * 4);
    float*    sbc   = (float*)carve((size_t)NBATCH * NOBS * 4);

    auto swz = [&](const float* s, _Float16* d, int K, int N, int Kp) {
        int total = (Kp / 32) * (N / 16) * 512;
        k_swizzle_W<<<dim3((total + 255) / 256), dim3(256), 0, stream>>>(s, d, K, N, Kp);
    };

    // ---- prep: weight swizzle, mean accumulator, padded inputs, obs constants ----
    swz(peW0, peW0s, 8, HID, 32);
    swz(peW1, peW1s, HID, HID, HID);
    swz(peW2, peW2s, HID, HID, HID);
    swz(hdW0, hdW0s, 2 * HID + COND, HID, 2 * HID + COND);
    swz(hdW1, hdW1s, HID, HID, HID);
    k_zero_f32<<<dim3((NBATCH * HID + 255) / 256), dim3(256), 0, stream>>>(sumpf, NBATCH * HID);
    k_build_in0<<<dim3(TOTAL_ROWS / 256), dim3(256), 0, stream>>>(coords, x_t, t_in, in0);
    k_obs_pre<<<dim3((NBATCH * NOBS + 255) / 256), dim3(256), 0, stream>>>(
        obs_coords, obs_mask, s4x, s4y, s4z, sbc);

    // ---- obs encoder (writes fragment-major obs_feat) ----
    k_obs_encoder<<<dim3(NBATCH * NOBS), dim3(COND), 0, stream>>>(
        obs_coords, obs_values, obs_mask, obs_ids, fe,
        oeW0, oeB0, oeW1, oeB1, oeW2, oeB2, ofswz);

    // ---- point encoder (3 WMMA GEMMs; last also accumulates column sums) ----
    dim3 gdim(TOTAL_ROWS / TM, HID / 128);
    k_gemm<0, true, false><<<gdim, dim3(256), 0, stream>>>(
        in0, nullptr, nullptr, peW0s, peB0, hA, nullptr, TOTAL_ROWS, HID, 32);
    k_gemm<0, true, false><<<gdim, dim3(256), 0, stream>>>(
        hA, nullptr, nullptr, peW1s, peB1, hB, nullptr, TOTAL_ROWS, HID, HID);
    k_gemm<0, false, true><<<gdim, dim3(256), 0, stream>>>(
        hB, nullptr, nullptr, peW2s, peB2, pf16, sumpf, TOTAL_ROWS, HID, HID);

    // ---- global encoder (mean via sumpf) ----
    k_global_mlp<<<dim3(NBATCH), dim3(HID), 0, stream>>>(sumpf, geW0, geB0, geW1, geB1, g16);

    // ---- aggregation: softmax stats then WMMA einsum with on-the-fly weights ----
    k_softmax_stats<<<dim3(TOTAL_ROWS / 256), dim3(256), 0, stream>>>(
        coords, s4x, s4y, s4z, sbc, rmax, rinv);
    k_lc_gemm<<<dim3(TOTAL_ROWS / 128), dim3(256), 0, stream>>>(
        coords, s4x, s4y, s4z, sbc, rmax, rinv, ofswz, lc16);

    // ---- head (concat fused into A-fragment source; 2 WMMA GEMMs + small out) ----
    k_gemm<1, true, false><<<gdim, dim3(256), 0, stream>>>(
        pf16, g16, lc16, hdW0s, hdB0, hA, nullptr, TOTAL_ROWS, HID, 2 * HID + COND);
    k_gemm<0, true, false><<<gdim, dim3(256), 0, stream>>>(
        hA, nullptr, nullptr, hdW1s, hdB1, hB, nullptr, TOTAL_ROWS, HID, HID);
    k_head_out<<<dim3(TOTAL_ROWS * NFIELDS / 256), dim3(256), 0, stream>>>(
        hB, hdW2, hdB2, out);
}
